// Seq2SeqNet_89756226552247
// MI455X (gfx1250) — compile-verified, matrix operand order checked by reading, not run on hardware
//
#include <hip/hip_runtime.h>
#include <math.h>

// ===========================================================================
// Seq2SeqNet forward for MI455X (gfx1250, wave32, WMMA).
//
// Input pointer layout assumption (d_in):
//   [0] q_geom_feats (512*1*14)   [1] k_geom_feats (512*512*14)
//   [2] msk (512*512)             [3] app_feats (512*1*315)
//   [4..121] params flattened in JAX pytree order (dict keys sorted):
//     jpus[0..3] x 21 leaves each:
//       k.W k.b ln_k.b ln_k.g ln_q.b ln_q.g o.W o.b q.W q.b
//       qb1_l.W qb1_l.b qb1_ln.b qb1_ln.g qb2_l.W qb2_l.b
//       qb2_ln.b qb2_ln.g scale v.W v.b
//     then k_full(4), k_geom(6), q_app(6), q_full(4), q_geom(6),
//     qf_l1(2), qf_l2(2), qf_ln1(2), qf_ln2(2)
// Output: (512,) float32 sigmoid scores.
// ===========================================================================

typedef __attribute__((ext_vector_type(16))) _Float16 v16h;
typedef __attribute__((ext_vector_type(8)))  float    v8f;

__device__ __forceinline__ float gelu_f(float x) {
  return 0.5f * x * (1.0f + erff(x * 0.70710678118654752440f));
}

__device__ __forceinline__ float wave_sum(float v) {
#pragma unroll
  for (int off = 16; off > 0; off >>= 1) v += __shfl_xor(v, off, 32);
  return v;
}

// ---------------------------------------------------------------------------
// Pack W (N x K, f32, row-major) into f16 WMMA B-fragment order:
//   Wp[((t*S + s)*32 + lane)*16 + e] = W[(t*16 + lane%16) * K + s*32 + kk]
// with kk = (e&7) + ((e>>3)<<4) + (lane>>4)*8  (CDNA5 16-bit frag layout,
// ISA 7.12.2). Out-of-range N/K padded with zeros, so the GEMM hot loop has
// no guards and loads each B fragment as one contiguous 32B per-lane read.
// ---------------------------------------------------------------------------
__global__ void pack_w_k(const float* __restrict__ W, _Float16* __restrict__ Wp,
                         int N, int K, int NT, int S) {
  int idx = blockIdx.x * blockDim.x + threadIdx.x;
  const int total = NT * S * 512;
  if (idx >= total) return;
  const int e  = idx & 15;
  const int l  = (idx >> 4) & 31;
  const int ts = idx >> 9;
  const int s  = ts % S;
  const int t  = ts / S;
  const int kk = (e & 7) + ((e >> 3) << 4) + ((l >> 4) << 3);
  const int n  = t * 16 + (l & 15);
  const int k  = s * 32 + kk;
  Wp[idx] = (_Float16)((n < N && k < K) ? W[n * K + k] : 0.f);
}

// ---------------------------------------------------------------------------
// Fused (LayerNorm ->) GEMM:  C = [res +] act(A @ W^T + bias)
// A = virtual concat [A | A2] along K (split==K, A2==A for no concat);
// A2 row = row & a2mask (broadcast of 512-row `ap` across 512x512 rows).
// LN uses precomputed per-row mean/rstd (`stats`), gamma/beta fused into the
// LDS staging (activations post-LN are O(1) -> f16-safe).
// Block = 64 rows x 128 cols = 8 waves; A staged once per 32-wide K step in
// fragment order; each wave: 1 contiguous B-frag load + 4 v_wmma per step.
// Requires: M % 64 == 0 (true for all call sites: 512 and 262144).
// ---------------------------------------------------------------------------
template <bool LN, bool ACT_GELU, bool RES>
__global__ __launch_bounds__(256) void wmma_gemm_k(
    const float* __restrict__ A, const float* __restrict__ A2, int split,
    int a2mask, const float* __restrict__ stats, const float* __restrict__ lng,
    const float* __restrict__ lnb, const _Float16* __restrict__ Wp,
    const float* __restrict__ bias, const float* __restrict__ res,
    float* __restrict__ C, int M, int N, int K, int NT, int S) {
  __shared__ _Float16 Af[4][32][16];  // 4 M-tiles, fragment order, 4KB
  const int tid   = threadIdx.x;
  const int wv    = tid >> 5;
  const int lane  = tid & 31;
  const int lo    = lane & 15;
  const int hi    = lane >> 4;
  const int row0  = blockIdx.y << 6;
  const int tileN = (blockIdx.x << 3) + wv;
  const int ncol  = (tileN << 4) + lo;
  const v16h* __restrict__ WpV = (const v16h*)Wp;

  v8f acc0 = {}, acc1 = {}, acc2 = {}, acc3 = {};

  for (int s = 0; s < S; ++s) {
    const int k0 = s << 5;
    __syncthreads();
    // Stage 64 rows x 32 K of LN-normalized A into LDS in fragment order.
#pragma unroll
    for (int q = 0; q < 8; ++q) {
      const int eidx = tid + (q << 8);
      const int r    = eidx >> 5;   // 0..63
      const int kk   = eidx & 31;
      const int row  = row0 + r;
      const int k    = k0 + kk;
      float x = 0.f;
      if (k < K) {
        x = (k < split) ? A[row * split + k]
                        : A2[(row & a2mask) * (K - split) + (k - split)];
        if (LN) x = (x - stats[2 * row]) * stats[2 * row + 1] * lng[k] + lnb[k];
      }
      const int lhi = (kk >> 3) & 1;
      const int le  = (kk & 7) + ((kk >> 4) << 3);
      Af[r >> 4][(r & 15) + (lhi << 4)][le] = (_Float16)x;
    }
    __syncthreads();
    if (tileN < NT) {  // wave-uniform: runtime EXEC stays all-ones for wmma
      const v16h b  = WpV[((tileN * S + s) << 5) + lane];
      const v16h a0 = *(const v16h*)&Af[0][lane][0];
      const v16h a1 = *(const v16h*)&Af[1][lane][0];
      const v16h a2 = *(const v16h*)&Af[2][lane][0];
      const v16h a3 = *(const v16h*)&Af[3][lane][0];
      acc0 = __builtin_amdgcn_wmma_f32_16x16x32_f16(false, a0, false, b,
                                                    (short)0, acc0, false, false);
      acc1 = __builtin_amdgcn_wmma_f32_16x16x32_f16(false, a1, false, b,
                                                    (short)0, acc1, false, false);
      acc2 = __builtin_amdgcn_wmma_f32_16x16x32_f16(false, a2, false, b,
                                                    (short)0, acc2, false, false);
      acc3 = __builtin_amdgcn_wmma_f32_16x16x32_f16(false, a3, false, b,
                                                    (short)0, acc3, false, false);
    }
  }

  if (ncol < N) {
    const float bvv = bias[ncol];
#pragma unroll
    for (int mt = 0; mt < 4; ++mt) {
      const v8f accv = (mt == 0) ? acc0 : (mt == 1) ? acc1 : (mt == 2) ? acc2
                                                                       : acc3;
#pragma unroll
      for (int r = 0; r < 8; ++r) {
        const int m = row0 + (mt << 4) + (hi << 3) + r;  // VGPR r -> M=r+8*hi
        float x = accv[r] + bvv;
        if (ACT_GELU) x = gelu_f(x);
        if (RES) x += res[m * N + ncol];
        C[m * N + ncol] = x;
      }
    }
  }
}

// Per-row mean / rstd over virtual concat row (for LN fusion into the GEMM).
__global__ __launch_bounds__(256) void rowstats_k(
    const float* __restrict__ A, const float* __restrict__ A2, int split,
    int a2mask, int M, int K, float* __restrict__ stats) {
  const int row  = blockIdx.x * 8 + (threadIdx.x >> 5);
  const int lane = threadIdx.x & 31;
  if (row >= M) return;
  float s = 0.f, ss = 0.f;
  for (int k = lane; k < K; k += 32) {
    const float x = (k < split) ? A[row * split + k]
                                : A2[(row & a2mask) * (K - split) + (k - split)];
    s += x; ss += x * x;
  }
  s = wave_sum(s); ss = wave_sum(ss);
  if (lane == 0) {
    const float m = s / (float)K;
    stats[2 * row]     = m;
    stats[2 * row + 1] = rsqrtf(ss / (float)K - m * m + 1e-6f);
  }
}

// Small / skinny-K linear (+gelu). Used for K=14 and K=315 input projections.
template <bool ACT_GELU>
__global__ void lin_small_k(const float* __restrict__ X,
                            const float* __restrict__ W,
                            const float* __restrict__ b, float* __restrict__ Y,
                            int M, int N, int K) {
  int idx = blockIdx.x * blockDim.x + threadIdx.x;
  const int total = M * N;
  for (; idx < total; idx += gridDim.x * blockDim.x) {
    const int m = idx / N, n = idx % N;
    float s = b[n];
    for (int k = 0; k < K; ++k) s += X[m * K + k] * W[n * K + k];
    Y[idx] = ACT_GELU ? gelu_f(s) : s;
  }
}

// Masked scores + softmax for one (n,h): s_j = (q.k_j)/8, mask, softmax.
__global__ __launch_bounds__(256) void attn_scores_k(
    const float* __restrict__ Q, const float* __restrict__ Kb,
    const float* __restrict__ msk, float* __restrict__ P) {
  __shared__ float qv[64];
  __shared__ float sb[512];
  __shared__ float red[256];
  const int n = blockIdx.x >> 1;
  const int h = blockIdx.x & 1;
  const int tid = threadIdx.x;
  if (tid < 64) qv[tid] = Q[n * 128 + h * 64 + tid];
  __syncthreads();
  for (int j = tid; j < 512; j += 256) {
    const float* kr = Kb + (n * 512 + j) * 128 + h * 64;
    float s = 0.f;
#pragma unroll 8
    for (int d = 0; d < 64; ++d) s += kr[d] * qv[d];
    s *= 0.125f;  // 1/sqrt(64)
    const float mk = msk[n * 512 + j];
    sb[j] = (mk == 0.f) ? -1.0e9f : s * mk;
  }
  __syncthreads();
  red[tid] = fmaxf(sb[tid], sb[tid + 256]);
  __syncthreads();
  for (int off = 128; off > 0; off >>= 1) {
    if (tid < off) red[tid] = fmaxf(red[tid], red[tid + off]);
    __syncthreads();
  }
  const float mx = red[0];
  __syncthreads();
  const float e0 = expf(sb[tid] - mx), e1 = expf(sb[tid + 256] - mx);
  red[tid] = e0 + e1;
  __syncthreads();
  for (int off = 128; off > 0; off >>= 1) {
    if (tid < off) red[tid] += red[tid + off];
    __syncthreads();
  }
  const float inv = 1.f / red[0];
  P[(n * 2 + h) * 512 + tid]       = e0 * inv;
  P[(n * 2 + h) * 512 + tid + 256] = e1 * inv;
}

// o[n, h*64+d] = sum_j P[n,h,j] * V[n,j, h*64+d]   (coalesced over t)
__global__ __launch_bounds__(128) void attn_out_k(const float* __restrict__ P,
                                                  const float* __restrict__ V,
                                                  float* __restrict__ O) {
  const int n = blockIdx.x;
  const int t = threadIdx.x;  // h = t>>6, d = t&63
  const float* pr = P + (n * 2 + (t >> 6)) * 512;
  float acc = 0.f;
  for (int j = 0; j < 512; ++j) acc += pr[j] * V[(n * 512 + j) * 128 + t];
  O[n * 128 + t] = acc;
}

// kf[i,j,c] += qa2[j,c] * scale[c]
__global__ void kf_update_k(float* __restrict__ kf,
                            const float* __restrict__ qa2,
                            const float* __restrict__ scale) {
  int idx = blockIdx.x * blockDim.x + threadIdx.x;
  const int total = 512 * 512 * 128;
  for (; idx < total; idx += gridDim.x * blockDim.x) {
    const int c = idx & 127;
    const int j = (idx >> 7) & 511;
    kf[idx] += qa2[j * 128 + c] * scale[c];
  }
}

// Final head: per row: sigmoid( ln(s,128) . W2 + b2 )
__global__ __launch_bounds__(256) void head_k(
    const float* __restrict__ S, const float* __restrict__ g,
    const float* __restrict__ be, const float* __restrict__ W2,
    const float* __restrict__ b2, float* __restrict__ out, int M) {
  const int row  = blockIdx.x * 8 + (threadIdx.x >> 5);
  const int lane = threadIdx.x & 31;
  if (row >= M) return;
  float s = 0.f, ss = 0.f;
  for (int k = lane; k < 128; k += 32) {
    const float x = S[row * 128 + k];
    s += x; ss += x * x;
  }
  s = wave_sum(s); ss = wave_sum(ss);
  const float m  = s / 128.f;
  const float rv = rsqrtf(ss / 128.f - m * m + 1e-6f);
  float acc = 0.f;
  for (int k = lane; k < 128; k += 32)
    acc += ((S[row * 128 + k] - m) * rv * g[k] + be[k]) * W2[k];
  acc = wave_sum(acc);
  if (lane == 0) out[row] = 1.f / (1.f + expf(-(acc + b2[0])));
}

// ===========================================================================
extern "C" void kernel_launch(void* const* d_in, const int* in_sizes, int n_in,
                              void* d_out, int out_size, void* d_ws,
                              size_t ws_size, hipStream_t stream) {
  (void)in_sizes; (void)n_in; (void)out_size; (void)ws_size;
  const int Nn = 512, R = 512 * 512;

  const float* qg  = (const float*)d_in[0];
  const float* kg  = (const float*)d_in[1];
  const float* msk = (const float*)d_in[2];
  const float* app = (const float*)d_in[3];
#define PRM(i) ((const float*)d_in[4 + (i)])
  const int KFULL = 84, KGEOM = 88, QAPP = 94, QFULL = 100, QGEO = 104;
  const int QFL1 = 110, QFL2 = 112, QFLN1 = 114, QFLN2 = 116;

  // ---- workspace arena (regions reused across phases; peak ~520 MB) ----
  float* ws     = (float*)d_ws;
  float* regA   = ws;                      // R*180: geom hidden, later K-proj
  float* regB   = regA + (size_t)R * 180;  // R*180: geom out,   later V-proj
  float* kf     = regB + (size_t)R * 180;  // R*128
  float* statsB = kf + (size_t)R * 128;    // 2*R
  float* p      = statsB + 2 * (size_t)R;
  float* qf0    = p;  p += 512 * 180;
  float* apb    = p;  p += 512 * 180;
  float* tmpS   = p;  p += 512 * 180;
  float* qfA    = p;  p += 512 * 256;
  float* qfB    = p;  p += 512 * 256;
  float* qa     = p;  p += 512 * 128;
  float* qproj  = p;  p += 512 * 128;
  float* qa2    = p;  p += 512 * 128;
  float* obuf   = p;  p += 512 * 128;
  float* probs  = p;  p += 512 * 2 * 512;
  float* statsS = p;  p += 2 * 512;
  float* sfin   = p;  p += 512 * 128;
  _Float16* wpack = (_Float16*)p;          // packed weights, <= 196KB

  auto stats = [&](const float* A, const float* A2, int split, int a2mask,
                   int M, int K, float* st) {
    rowstats_k<<<(M + 7) / 8, 256, 0, stream>>>(A, A2, split, a2mask, M, K, st);
  };
  auto gemm = [&](bool ln, bool act, bool resid, const float* A,
                  const float* A2, int split, int a2mask, const float* st,
                  const float* g, const float* b, const float* W,
                  const float* bias, const float* res, float* C, int M, int N,
                  int K) {
    const int NT = (N + 15) / 16, S = (K + 31) / 32;
    pack_w_k<<<(NT * S * 512 + 255) / 256, 256, 0, stream>>>(W, wpack, N, K,
                                                             NT, S);
    dim3 grid((NT + 7) / 8, M >> 6);
    if (ln && act && !resid)
      wmma_gemm_k<true, true, false><<<grid, 256, 0, stream>>>(
          A, A2, split, a2mask, st, g, b, wpack, bias, res, C, M, N, K, NT, S);
    else if (ln && !act && !resid)
      wmma_gemm_k<true, false, false><<<grid, 256, 0, stream>>>(
          A, A2, split, a2mask, st, g, b, wpack, bias, res, C, M, N, K, NT, S);
    else if (ln && act && resid)
      wmma_gemm_k<true, true, true><<<grid, 256, 0, stream>>>(
          A, A2, split, a2mask, st, g, b, wpack, bias, res, C, M, N, K, NT, S);
    else
      wmma_gemm_k<false, true, true><<<grid, 256, 0, stream>>>(
          A, A2, split, a2mask, st, g, b, wpack, bias, res, C, M, N, K, NT, S);
  };

  // ---- q-path geom MLP: (512,14) -> gelu l1 -> ln -> gelu l2 -> qf0 ----
  lin_small_k<true><<<(512 * 180 + 255) / 256, 256, 0, stream>>>(
      qg, PRM(QGEO + 0), PRM(QGEO + 1), tmpS, 512, 180, 14);
  stats(tmpS, tmpS, 180, 0, 512, 180, statsS);
  gemm(true, true, false, tmpS, tmpS, 180, 0, statsS, PRM(QGEO + 5),
       PRM(QGEO + 4), PRM(QGEO + 2), PRM(QGEO + 3), nullptr, qf0, 512, 180, 180);

  // ---- appearance MLP: (512,315) -> apb ----
  lin_small_k<true><<<(512 * 180 + 255) / 256, 256, 0, stream>>>(
      app, PRM(QAPP + 0), PRM(QAPP + 1), tmpS, 512, 180, 315);
  stats(tmpS, tmpS, 180, 0, 512, 180, statsS);
  gemm(true, true, false, tmpS, tmpS, 180, 0, statsS, PRM(QAPP + 5),
       PRM(QAPP + 4), PRM(QAPP + 2), PRM(QAPP + 3), nullptr, apb, 512, 180, 180);

  // ---- k-path geom MLP: (262144,14) -> regA -> ln-fused WMMA l2 -> regB ----
  lin_small_k<true><<<(R * 180 + 255) / 256, 256, 0, stream>>>(
      kg, PRM(KGEOM + 0), PRM(KGEOM + 1), regA, R, 180, 14);
  stats(regA, regA, 180, 0, R, 180, statsB);
  gemm(true, true, false, regA, regA, 180, 0, statsB, PRM(KGEOM + 5),
       PRM(KGEOM + 4), PRM(KGEOM + 2), PRM(KGEOM + 3), nullptr, regB, R, 180,
       180);

  // ---- q_full: ln(concat[qf0|apb]) -> gelu linear -> qfA (512,256) ----
  stats(qf0, apb, 180, 511, 512, 360, statsS);
  gemm(true, true, false, qf0, apb, 180, 511, statsS, PRM(QFULL + 3),
       PRM(QFULL + 2), PRM(QFULL + 0), PRM(QFULL + 1), nullptr, qfA, 512, 256,
       360);

  // ---- k_full: ln(concat[regB | broadcast apb]) -> gelu -> kf (R,128) ----
  stats(regB, apb, 180, 511, R, 360, statsB);
  gemm(true, true, false, regB, apb, 180, 511, statsB, PRM(KFULL + 3),
       PRM(KFULL + 2), PRM(KFULL + 0), PRM(KFULL + 1), nullptr, kf, R, 128, 360);

  // ---- JPU stack ----
  float* qfC = qfA;
  float* qfN = qfB;
  for (int i = 0; i < 4; ++i) {
    const int jb = i * 21;
    // qa = gelu(qb1_l(ln(qf)))
    stats(qfC, qfC, 256, 0, 512, 256, statsS);
    gemm(true, true, false, qfC, qfC, 256, 0, statsS, PRM(jb + 13),
         PRM(jb + 12), PRM(jb + 10), PRM(jb + 11), nullptr, qa, 512, 128, 256);
    // kin/qin LN stats
    stats(kf, kf, 128, 0, R, 128, statsB);
    stats(qa, qa, 128, 0, 512, 128, statsS);
    // q / K / V projections (LN fused)
    gemm(true, false, false, qa, qa, 128, 0, statsS, PRM(jb + 5), PRM(jb + 4),
         PRM(jb + 8), PRM(jb + 9), nullptr, qproj, 512, 128, 128);
    gemm(true, false, false, kf, kf, 128, 0, statsB, PRM(jb + 3), PRM(jb + 2),
         PRM(jb + 0), PRM(jb + 1), nullptr, regA, R, 128, 128);
    gemm(true, false, false, kf, kf, 128, 0, statsB, PRM(jb + 3), PRM(jb + 2),
         PRM(jb + 19), PRM(jb + 20), nullptr, regB, R, 128, 128);
    // masked softmax attention (M=1 per query -> VALU matvec path)
    attn_scores_k<<<1024, 256, 0, stream>>>(qproj, regA, msk, probs);
    attn_out_k<<<512, 128, 0, stream>>>(probs, regB, obuf);
    // qa2 = qa + gelu(o_proj(obuf))
    gemm(false, true, true, obuf, obuf, 128, 0, nullptr, nullptr, nullptr,
         PRM(jb + 6), PRM(jb + 7), qa, qa2, 512, 128, 128);
    // kf += broadcast(qa2 * scale)
    kf_update_k<<<(R * 128 + 255) / 256, 256, 0, stream>>>(kf, qa2,
                                                           PRM(jb + 18));
    // qf = qf + gelu(qb2_l(ln(qa2)))
    stats(qa2, qa2, 128, 0, 512, 128, statsS);
    gemm(true, true, true, qa2, qa2, 128, 0, statsS, PRM(jb + 17),
         PRM(jb + 16), PRM(jb + 14), PRM(jb + 15), qfC, qfN, 512, 256, 128);
    float* t = qfC; qfC = qfN; qfN = t;
  }

  // ---- head: s = gelu(qf_l1(ln(qf))); out = sigmoid(qf_l2(ln(s))) ----
  stats(qfC, qfC, 256, 0, 512, 256, statsS);
  gemm(true, true, false, qfC, qfC, 256, 0, statsS, PRM(QFLN1 + 1),
       PRM(QFLN1 + 0), PRM(QFL1 + 0), PRM(QFL1 + 1), nullptr, sfin, 512, 128,
       256);
  head_k<<<(Nn + 7) / 8, 256, 0, stream>>>(sfin, PRM(QFLN2 + 1), PRM(QFLN2 + 0),
                                           PRM(QFL2 + 0), PRM(QFL2 + 1),
                                           (float*)d_out, Nn);
#undef PRM
}
